// EBT_9706626090101
// MI455X (gfx1250) — matrix operand order repaired and checked by reading, not compiled, not run on hardware
//
#include <hip/hip_runtime.h>

// ---------------------------------------------------------------------------
// Types
// ---------------------------------------------------------------------------
typedef __bf16 bf16_t;
typedef __bf16 v8bf  __attribute__((ext_vector_type(8)));
typedef __bf16 v16bf __attribute__((ext_vector_type(16)));
typedef float  v8f   __attribute__((ext_vector_type(8)));
typedef unsigned int u32x4 __attribute__((ext_vector_type(4)));
typedef int          i32x4 __attribute__((ext_vector_type(4)));
typedef int          i32x8 __attribute__((ext_vector_type(8)));

#define BB    32
#define SEQ   512
#define DIM   768
#define NH    12
#define HD    64
#define HIDD  3072
#define NL    12
#define MM    (BB * SEQ)        // 16384 rows
#define QKVS  (3 * DIM)         // 2304

#define EPI_STORE 0
#define EPI_GELU  1
#define EPI_GATE  2

static __device__ __forceinline__ bf16_t f2bf(float f) {
  unsigned u = __float_as_uint(f);
  u += 0x7fffu + ((u >> 16) & 1u);          // round-to-nearest-even
  unsigned short s = (unsigned short)(u >> 16);
  bf16_t r;
  __builtin_memcpy(&r, &s, 2);
  return r;
}

static __device__ __forceinline__ v16bf combine16(v8bf lo, v8bf hi) {
  return __builtin_shufflevector(lo, hi, 0, 1, 2, 3, 4, 5, 6, 7,
                                          8, 9, 10, 11, 12, 13, 14, 15);
}

static __device__ __forceinline__ v8f wmma_bf16(v16bf a, v16bf b, v8f c) {
  // D = A(16x32 bf16) * B(32x16 bf16) + C(16x16 f32)
  return __builtin_amdgcn_wmma_f32_16x16x32_bf16(false, a, false, b,
                                                 (short)0, c, false, false);
}

static __device__ __forceinline__ float gelu_tanh(float x) {
  return 0.5f * x * (1.0f + tanhf(0.7978845608028654f * (x + 0.044715f * x * x * x)));
}

// ---------------------------------------------------------------------------
// Tensor Data Mover: DMA a 64-row x 32-element bf16 tile (row stride K elems)
// from global memory into LDS at lds_off, padding each 64B row to 80B so the
// LDS image matches a [64][40] bf16 array (16B-aligned WMMA fragments).
// D# bitfields per CDNA5 ISA.
// This toolchain's builtin takes 6 args: (g0, g1, g2, g3, g_extra, cpol).
// ---------------------------------------------------------------------------
static __device__ __forceinline__ void tdm_load_tile_64x32(
    const bf16_t* gsrc, unsigned lds_off, int stride_elems) {
  unsigned long long ga = (unsigned long long)(uintptr_t)gsrc;
  u32x4 g0;
  g0[0] = 1u;                                     // count=1, user descriptor
  g0[1] = lds_off;                                // lds_addr (bytes)
  g0[2] = (unsigned)ga;                           // global_addr[31:0]
  g0[3] = (unsigned)((ga >> 32) & 0x01FFFFFFu)    // global_addr[56:32]
          | 0x80000000u;                          // type=2 ("image")
  i32x8 g1;
  g1[0] = (1 << 16)      // data_size = 2 bytes
        | (1 << 20)      // pad_enable
        | (3 << 22)      // pad_interval: 16 DWORDs (64B) between pads
        | (3 << 25);     // pad_amount: 4 DWORDs (16B)
  g1[1] = (32 << 16);    // tensor_dim0 = 32 elements (bits 79:48, low half)
  g1[2] = (64 << 16);    // tensor_dim1 = 64 rows    (bits 111:80, low half)
  g1[3] = (32 << 16);    // tile_dim0 = 32 elements  (bits 127:112)
  g1[4] = 64;            // tile_dim1 = 64 rows      (bits 143:128)
  g1[5] = stride_elems;  // tensor_dim0_stride[31:0] (bits 207:160)
  g1[6] = 0;
  g1[7] = 0;
  i32x4 z4 = {0, 0, 0, 0};
  i32x8 z8 = {0, 0, 0, 0, 0, 0, 0, 0};
  __builtin_amdgcn_tensor_load_to_lds(g0, g1, z4, z4, z8, 0);
}

// ---------------------------------------------------------------------------
// c = ds_table[dataset] + sg_table[spacegroup];  store silu(c)   [B, D] f32
// ---------------------------------------------------------------------------
__global__ __launch_bounds__(256)
void cond_silu_kernel(const float* __restrict__ ds_t, const float* __restrict__ sg_t,
                      const int* __restrict__ dataset, const int* __restrict__ spacegroup,
                      float* __restrict__ csilu) {
  const int b = blockIdx.x;
  const float* a0 = ds_t + (size_t)dataset[b] * DIM;
  const float* a1 = sg_t + (size_t)spacegroup[b] * DIM;
  for (int d = threadIdx.x; d < DIM; d += 256) {
    float c = a0[d] + a1[d];
    csilu[(size_t)b * DIM + d] = c / (1.0f + __expf(-c));
  }
}

// ---------------------------------------------------------------------------
// h = x @ W_xemb + b_xemb + pos_embedding(n)      [M, D] f32
// ---------------------------------------------------------------------------
__global__ __launch_bounds__(256)
void embed_kernel(const float* __restrict__ x, const float* __restrict__ Wx,
                  const float* __restrict__ bx, float* __restrict__ h) {
  const int row = blockIdx.x;
  const int n = row & (SEQ - 1);
  __shared__ float xs[16];
  if (threadIdx.x < 16) xs[threadIdx.x] = x[(size_t)row * 16 + threadIdx.x];
  __syncthreads();
  for (int d = threadIdx.x; d < DIM; d += 256) {
    float acc = bx[d];
#pragma unroll
    for (int k = 0; k < 16; ++k) acc += xs[k] * Wx[k * DIM + d];
    int K = (d < (DIM / 2)) ? d : d - (DIM / 2);
    float ang = (float)n * 3.14159265358979f /
                __powf(2048.0f, 2.0f * (float)K / (float)DIM);
    acc += (d < (DIM / 2)) ? __sinf(ang) : __cosf(ang);
    h[(size_t)row * DIM + d] = acc;
  }
}

// ---------------------------------------------------------------------------
// mods = csilu @ W[D, NO] + bias[NO]   (small M=32 GEMM, scalar fp32)
// ---------------------------------------------------------------------------
__global__ __launch_bounds__(256)
void ada_kernel(const float* __restrict__ cs, const float* __restrict__ W,
                const float* __restrict__ bias, float* __restrict__ out, int NO) {
  const int j = blockIdx.x * 256 + threadIdx.x;
  const int b = blockIdx.y;
  if (j >= NO) return;
  const float* c = cs + (size_t)b * DIM;
  float acc = bias[j];
  for (int k = 0; k < DIM; ++k) acc += c[k] * W[(size_t)k * NO + j];
  out[(size_t)b * NO + j] = acc;
}

// ---------------------------------------------------------------------------
// Transpose-convert: W[K,N] f32 -> Wt[N,K] bf16  (tiled through LDS).
// Pre-transposed weights make the GEMM B-tile a contiguous rows-of-32 load,
// so the TDM can stage both A and B tiles with the same descriptor shape.
// ---------------------------------------------------------------------------
__global__ __launch_bounds__(256)
void cvtT_kernel(const float* __restrict__ W, bf16_t* __restrict__ Wt, int K, int N) {
  __shared__ float t[32][33];
  const int nb = blockIdx.x * 32, kb = blockIdx.y * 32;
  const int tx = threadIdx.x, ty = threadIdx.y;   // 32 x 8
#pragma unroll
  for (int i = 0; i < 4; ++i)
    t[ty + i * 8][tx] = W[(size_t)(kb + ty + i * 8) * N + nb + tx];
  __syncthreads();
#pragma unroll
  for (int i = 0; i < 4; ++i)
    Wt[(size_t)(nb + ty + i * 8) * K + kb + tx] = f2bf(t[tx][ty + i * 8]);
}

// ---------------------------------------------------------------------------
// out_bf16 = modulate(layernorm(h), mods[sh], mods[sc])     one block per row
// ---------------------------------------------------------------------------
__global__ __launch_bounds__(256)
void ln_mod_kernel(const float* __restrict__ h, const float* __restrict__ mods,
                   int sh_off, int sc_off, int mstride, bf16_t* __restrict__ out) {
  const int row = blockIdx.x, b = row >> 9;  // 512 rows per batch
  const int tid = threadIdx.x;
  const float* x = h + (size_t)row * DIM;
  float v0 = x[tid], v1 = x[tid + 256], v2 = x[tid + 512];
  float s = v0 + v1 + v2;
  float q = v0 * v0 + v1 * v1 + v2 * v2;
  __shared__ float r1[8], r2[8];
#pragma unroll
  for (int off = 16; off >= 1; off >>= 1) {
    s += __shfl_xor(s, off, 32);
    q += __shfl_xor(q, off, 32);
  }
  if ((tid & 31) == 0) { r1[tid >> 5] = s; r2[tid >> 5] = q; }
  __syncthreads();
  float S = 0.f, Q = 0.f;
#pragma unroll
  for (int i = 0; i < 8; ++i) { S += r1[i]; Q += r2[i]; }
  const float mean = S * (1.0f / (float)DIM);
  const float var  = Q * (1.0f / (float)DIM) - mean * mean;
  const float rinv = rsqrtf(var + 1e-6f);
  const float* mr = mods + (size_t)b * mstride;
  bf16_t* o = out + (size_t)row * DIM;
#pragma unroll
  for (int i = 0; i < 3; ++i) {
    int d = tid + i * 256;
    float v = (i == 0) ? v0 : ((i == 1) ? v1 : v2);
    float y = (v - mean) * rinv;
    y = y * (1.0f + mr[sc_off + d]) + mr[sh_off + d];
    o[d] = f2bf(y);
  }
}

// ---------------------------------------------------------------------------
// bf16 WMMA GEMM with TDM double-buffered staging:
//   C[M,N] = A[M,K] @ Bt[N,K]^T + bias   (+ epilogue)
//   64x64 tile / block of 256 threads (8 waves), K-step 32.
//   Wave 0 DMAs tile k+1 into the alternate LDS buffer while all waves run
//   WMMAs on tile k; s_wait_tensorcnt + barrier flips buffers.
// ---------------------------------------------------------------------------
__global__ __launch_bounds__(256)
void gemm_bf16_kernel(const bf16_t* __restrict__ A, const bf16_t* __restrict__ Bt,
                      const float* __restrict__ bias, bf16_t* __restrict__ Cbf,
                      float* __restrict__ Hres, const float* __restrict__ gate,
                      int K, int N, int gate_stride, int epi) {
  __shared__ bf16_t As[2][64][40];   // [m][k], 80B row stride (TDM pad)
  __shared__ bf16_t Bs[2][64][40];   // [n][k]
  const int tid  = threadIdx.x;
  const int wave = tid >> 5, lane = tid & 31;
  const int l15 = lane & 15, hsel = lane >> 4;
  const int mblk = blockIdx.y * 64, nblk = blockIdx.x * 64;
  const int wm = (wave & 3) * 16, wn = (wave >> 2) * 32;
  v8f acc0 = {}, acc1 = {};

  const bf16_t* aptr = A + (size_t)mblk * K;     // tile start (advances in k)
  const bf16_t* bptr = Bt + (size_t)nblk * K;
  // Flat->LDS-offset: LDS aperture keeps the byte offset in addr[31:0].
  const unsigned asb = (unsigned)(uintptr_t)&As[0][0][0];
  const unsigned bsb = (unsigned)(uintptr_t)&Bs[0][0][0];
  const unsigned BUFB = 64 * 40 * 2;             // 5120B per buffer

  if (wave == 0) {
    tdm_load_tile_64x32(aptr, asb, K);
    tdm_load_tile_64x32(bptr, bsb, K);
    __builtin_amdgcn_s_wait_tensorcnt(0);
  }
  __syncthreads();

  int buf = 0;
  for (int k0 = 0; k0 < K; k0 += 32) {
    if (wave == 0 && (k0 + 32) < K) {            // prefetch next tile via TDM
      tdm_load_tile_64x32(aptr + k0 + 32, asb + (buf ^ 1) * BUFB, K);
      tdm_load_tile_64x32(bptr + k0 + 32, bsb + (buf ^ 1) * BUFB, K);
    }
    // A fragment (16x32): lanes<16 rows m, K 0-15; lanes>=16: K 16-31 (contiguous)
    v8bf alo = *reinterpret_cast<const v8bf*>(&As[buf][wm + l15][hsel * 16]);
    v8bf ahi = *reinterpret_cast<const v8bf*>(&As[buf][wm + l15][hsel * 16 + 8]);
    v16bf af = combine16(alo, ahi);
    // B fragment (32x16): lane = column n, halves = contiguous K 0-15 / 16-31
    v8bf b0lo = *reinterpret_cast<const v8bf*>(&Bs[buf][wn + l15][hsel * 16]);
    v8bf b0hi = *reinterpret_cast<const v8bf*>(&Bs[buf][wn + l15][hsel * 16 + 8]);
    v8bf b1lo = *reinterpret_cast<const v8bf*>(&Bs[buf][wn + 16 + l15][hsel * 16]);
    v8bf b1hi = *reinterpret_cast<const v8bf*>(&Bs[buf][wn + 16 + l15][hsel * 16 + 8]);
    acc0 = wmma_bf16(af, combine16(b0lo, b0hi), acc0);
    acc1 = wmma_bf16(af, combine16(b1lo, b1hi), acc1);
    if (wave == 0) __builtin_amdgcn_s_wait_tensorcnt(0);
    __syncthreads();
    buf ^= 1;
  }

  // C layout: vgpr r -> (M = r + 8*hsel, N = l15) within 16x16 tile
  const int mrow = mblk + wm + hsel * 8;
  const int c0 = nblk + wn + l15, c1 = c0 + 16;
  const float bia0 = bias[c0], bia1 = bias[c1];
#pragma unroll
  for (int r = 0; r < 8; ++r) {
    const int m = mrow + r;
    float v0 = acc0[r] + bia0;
    float v1 = acc1[r] + bia1;
    if (epi == EPI_STORE) {
      Cbf[(size_t)m * N + c0] = f2bf(v0);
      Cbf[(size_t)m * N + c1] = f2bf(v1);
    } else if (epi == EPI_GELU) {
      Cbf[(size_t)m * N + c0] = f2bf(gelu_tanh(v0));
      Cbf[(size_t)m * N + c1] = f2bf(gelu_tanh(v1));
    } else {  // EPI_GATE: h += gate[b,:] * (acc + bias)
      const int b = m >> 9;
      const float* g = gate + (size_t)b * gate_stride;
      Hres[(size_t)m * N + c0] += g[c0] * v0;
      Hres[(size_t)m * N + c1] += g[c1] * v1;
    }
  }
}

// ---------------------------------------------------------------------------
// Flash attention, one block per (b*head, 128-row Q tile), 8 waves x 16 rows.
// qkv layout: [M, 3, NH, HD] bf16.   masked key (mask!=0) -> -inf score.
// ---------------------------------------------------------------------------
__global__ __launch_bounds__(256)
void attention_kernel(const bf16_t* __restrict__ qkv,
                      const unsigned char* __restrict__ mask,
                      bf16_t* __restrict__ Obuf) {
  const int bh = blockIdx.x;
  const int b = bh / NH, hh = bh % NH;
  const int tid = threadIdx.x;
  const int wave = tid >> 5, lane = tid & 31;
  const int l15 = lane & 15, hsel = lane >> 4;
  const int mrow = b * SEQ + blockIdx.y * 128 + wave * 16;  // wave's first q row

  __shared__ bf16_t Vt[64][40];       // V^T for current 32-key tile: [d][key]
  __shared__ bf16_t Pw[8][16][40];    // per-wave P tile: [qrow][key]

  // Q fragments (held for whole kernel): head-dim split into 2 x K=32 frags
  const bf16_t* qrow = qkv + (size_t)(mrow + l15) * QKVS + hh * HD;
  v16bf qf0 = combine16(*reinterpret_cast<const v8bf*>(qrow + hsel * 16),
                        *reinterpret_cast<const v8bf*>(qrow + hsel * 16 + 8));
  v16bf qf1 = combine16(*reinterpret_cast<const v8bf*>(qrow + 32 + hsel * 16),
                        *reinterpret_cast<const v8bf*>(qrow + 40 + hsel * 16));

  v8f oacc[4];
  v8f zero = {};
#pragma unroll
  for (int i = 0; i < 4; ++i) oacc[i] = zero;
  float mrun[8], lrun[8];
#pragma unroll
  for (int r = 0; r < 8; ++r) { mrun[r] = -1e30f; lrun[r] = 0.0f; }

  const int kvrow0 = b * SEQ;
  const int stg_key = tid & 31, stg_d = (tid >> 5) * 8;

  for (int kt = 0; kt < SEQ; kt += 32) {
    // cooperatively stage V^T (32 keys x 64 d -> Vt[d][key])
    {
      const bf16_t* vr = qkv + (size_t)(kvrow0 + kt + stg_key) * QKVS +
                         (2 * NH + hh) * HD + stg_d;
      v8bf vv = *reinterpret_cast<const v8bf*>(vr);
#pragma unroll
      for (int i = 0; i < 8; ++i) Vt[stg_d + i][stg_key] = vv[i];
    }
    __syncthreads();

    // S = Q @ K^T : K rows are contiguous in d == native B-fragment layout
    const bf16_t* kr0 = qkv + (size_t)(kvrow0 + kt + l15) * QKVS +
                        (NH + hh) * HD + hsel * 16;
    const bf16_t* kr1 = kr0 + (size_t)16 * QKVS;
    v8f s0 = {}, s1 = {};
    s0 = wmma_bf16(qf0, combine16(*reinterpret_cast<const v8bf*>(kr0),
                                  *reinterpret_cast<const v8bf*>(kr0 + 8)), s0);
    s0 = wmma_bf16(qf1, combine16(*reinterpret_cast<const v8bf*>(kr0 + 32),
                                  *reinterpret_cast<const v8bf*>(kr0 + 40)), s0);
    s1 = wmma_bf16(qf0, combine16(*reinterpret_cast<const v8bf*>(kr1),
                                  *reinterpret_cast<const v8bf*>(kr1 + 8)), s1);
    s1 = wmma_bf16(qf1, combine16(*reinterpret_cast<const v8bf*>(kr1 + 32),
                                  *reinterpret_cast<const v8bf*>(kr1 + 40)), s1);

    const bool mk0 = mask[b * SEQ + kt + l15] != 0;
    const bool mk1 = mask[b * SEQ + kt + 16 + l15] != 0;

    float tmax[8];
#pragma unroll
    for (int r = 0; r < 8; ++r) {
      float a0 = mk0 ? -1e30f : s0[r] * 0.125f;   // 1/sqrt(64)
      float a1 = mk1 ? -1e30f : s1[r] * 0.125f;
      s0[r] = a0; s1[r] = a1;
      tmax[r] = fmaxf(a0, a1);
    }
    // row max across the 16 lanes of each half-wave (xor masks stay in-half)
#pragma unroll
    for (int r = 0; r < 8; ++r) {
#pragma unroll
      for (int off = 8; off >= 1; off >>= 1)
        tmax[r] = fmaxf(tmax[r], __shfl_xor(tmax[r], off, 32));
    }

#pragma unroll
    for (int r = 0; r < 8; ++r) {
      float mn = fmaxf(mrun[r], tmax[r]);
      float sc = __expf(mrun[r] - mn);
      mrun[r] = mn;
      float p0 = __expf(s0[r] - mn);
      float p1 = __expf(s1[r] - mn);
      float ts = p0 + p1;
#pragma unroll
      for (int off = 8; off >= 1; off >>= 1) ts += __shfl_xor(ts, off, 32);
      lrun[r] = lrun[r] * sc + ts;
#pragma unroll
      for (int t = 0; t < 4; ++t) oacc[t][r] *= sc;
      int prow = hsel * 8 + r;                       // C-layout row -> P row
      Pw[wave][prow][l15]      = f2bf(p0);
      Pw[wave][prow][16 + l15] = f2bf(p1);
    }

    // O += P @ V : P from LDS in A-fragment layout, V^T contiguous per lane
    v8bf plo = *reinterpret_cast<const v8bf*>(&Pw[wave][l15][hsel * 16]);
    v8bf phi = *reinterpret_cast<const v8bf*>(&Pw[wave][l15][hsel * 16 + 8]);
    v16bf pf = combine16(plo, phi);
#pragma unroll
    for (int dt = 0; dt < 4; ++dt) {
      v8bf vlo = *reinterpret_cast<const v8bf*>(&Vt[dt * 16 + l15][hsel * 16]);
      v8bf vhi = *reinterpret_cast<const v8bf*>(&Vt[dt * 16 + l15][hsel * 16 + 8]);
      oacc[dt] = wmma_bf16(pf, combine16(vlo, vhi), oacc[dt]);
    }
    __syncthreads();   // before next tile overwrites Vt
  }

#pragma unroll
  for (int r = 0; r < 8; ++r) {
    const float inv = 1.0f / lrun[r];
    const int m = mrow + hsel * 8 + r;
#pragma unroll
    for (int dt = 0; dt < 4; ++dt)
      Obuf[(size_t)m * DIM + hh * HD + dt * 16 + l15] = f2bf(oacc[dt][r] * inv);
  }
}

// ---------------------------------------------------------------------------
// out[row] = modulate(layernorm(h), mods_f) . W_out
// ---------------------------------------------------------------------------
__global__ __launch_bounds__(256)
void final_kernel(const float* __restrict__ h, const float* __restrict__ modsf,
                  const float* __restrict__ Wout, float* __restrict__ out) {
  const int row = blockIdx.x, b = row >> 9;
  const int tid = threadIdx.x;
  const float* x = h + (size_t)row * DIM;
  float v0 = x[tid], v1 = x[tid + 256], v2 = x[tid + 512];
  float s = v0 + v1 + v2;
  float q = v0 * v0 + v1 * v1 + v2 * v2;
  __shared__ float r1[8], r2[8], r3[8];
#pragma unroll
  for (int off = 16; off >= 1; off >>= 1) {
    s += __shfl_xor(s, off, 32);
    q += __shfl_xor(q, off, 32);
  }
  if ((tid & 31) == 0) { r1[tid >> 5] = s; r2[tid >> 5] = q; }
  __syncthreads();
  float S = 0.f, Q = 0.f;
#pragma unroll
  for (int i = 0; i < 8; ++i) { S += r1[i]; Q += r2[i]; }
  const float mean = S * (1.0f / (float)DIM);
  const float var  = Q * (1.0f / (float)DIM) - mean * mean;
  const float rinv = rsqrtf(var + 1e-6f);
  const float* mr = modsf + (size_t)b * (2 * DIM);
  float acc = 0.f;
#pragma unroll
  for (int i = 0; i < 3; ++i) {
    int d = tid + i * 256;
    float v = (i == 0) ? v0 : ((i == 1) ? v1 : v2);
    float y = (v - mean) * rinv;
    y = y * (1.0f + mr[DIM + d]) + mr[d];
    acc += y * Wout[d];
  }
#pragma unroll
  for (int off = 16; off >= 1; off >>= 1) acc += __shfl_xor(acc, off, 32);
  if ((tid & 31) == 0) r3[tid >> 5] = acc;
  __syncthreads();
  if (tid == 0) {
    float T = 0.f;
#pragma unroll
    for (int i = 0; i < 8; ++i) T += r3[i];
    out[row] = T;
  }
}

// ---------------------------------------------------------------------------
// Host orchestration
// ---------------------------------------------------------------------------
extern "C" void kernel_launch(void* const* d_in, const int* in_sizes, int n_in,
                              void* d_out, int out_size, void* d_ws, size_t ws_size,
                              hipStream_t stream) {
  (void)in_sizes; (void)n_in; (void)out_size; (void)ws_size;
  const float* x       = (const float*)d_in[0];
  const float* W_xemb  = (const float*)d_in[1];
  const float* b_xemb  = (const float*)d_in[2];
  const float* ds_tab  = (const float*)d_in[3];
  const float* sg_tab  = (const float*)d_in[4];
  const float* Wqkv    = (const float*)d_in[5];
  const float* bqkv    = (const float*)d_in[6];
  const float* Wo      = (const float*)d_in[7];
  const float* bo      = (const float*)d_in[8];
  const float* W1      = (const float*)d_in[9];
  const float* b1      = (const float*)d_in[10];
  const float* W2      = (const float*)d_in[11];
  const float* b2      = (const float*)d_in[12];
  const float* Wada    = (const float*)d_in[13];
  const float* bada    = (const float*)d_in[14];
  const float* Wada_f  = (const float*)d_in[15];
  const float* bada_f  = (const float*)d_in[16];
  const float* W_out   = (const float*)d_in[17];
  const int*   dataset = (const int*)d_in[18];
  const int*   spacegr = (const int*)d_in[19];
  const unsigned char* mask = (const unsigned char*)d_in[20];
  float* out = (float*)d_out;

  char* ws = (char*)d_ws;
  size_t off = 0;
  auto salloc = [&](size_t bytes) {
    size_t o = off;
    off += (bytes + 255) & ~(size_t)255;
    return o;
  };
  float*  h      = (float*)(ws + salloc((size_t)MM * DIM * 4));
  float*  csilu  = (float*)(ws + salloc((size_t)BB * DIM * 4));
  float*  mods   = (float*)(ws + salloc((size_t)BB * 6 * DIM * 4));
  float*  modsf  = (float*)(ws + salloc((size_t)BB * 2 * DIM * 4));
  bf16_t* xm     = (bf16_t*)(ws + salloc((size_t)MM * DIM * 2));
  bf16_t* qkvb   = (bf16_t*)(ws + salloc((size_t)MM * QKVS * 2));
  bf16_t* attnb  = (bf16_t*)(ws + salloc((size_t)MM * DIM * 2));
  bf16_t* hidb   = (bf16_t*)(ws + salloc((size_t)MM * HIDD * 2));
  bf16_t* wq_bf  = (bf16_t*)(ws + salloc((size_t)DIM * QKVS * 2));   // [N,K]
  bf16_t* wo_bf  = (bf16_t*)(ws + salloc((size_t)DIM * DIM * 2));    // [N,K]
  bf16_t* w1_bf  = (bf16_t*)(ws + salloc((size_t)DIM * HIDD * 2));   // [N,K]
  bf16_t* w2_bf  = (bf16_t*)(ws + salloc((size_t)HIDD * DIM * 2));   // [N,K]

  dim3 blk(256);
  dim3 tblk(32, 8);

  cond_silu_kernel<<<BB, blk, 0, stream>>>(ds_tab, sg_tab, dataset, spacegr, csilu);
  embed_kernel<<<MM, blk, 0, stream>>>(x, W_xemb, b_xemb, h);

  // transpose-convert a [K,N] f32 weight into [N,K] bf16
  auto cvtT = [&](const float* s, bf16_t* d, int K, int N) {
    cvtT_kernel<<<dim3(N / 32, K / 32), tblk, 0, stream>>>(s, d, K, N);
  };

  for (int l = 0; l < NL; ++l) {
    ada_kernel<<<dim3((6 * DIM) / 256, BB), blk, 0, stream>>>(
        csilu, Wada + (size_t)l * DIM * 6 * DIM, bada + (size_t)l * 6 * DIM,
        mods, 6 * DIM);

    cvtT(Wqkv + (size_t)l * DIM * QKVS, wq_bf, DIM, QKVS);
    cvtT(Wo   + (size_t)l * DIM * DIM,  wo_bf, DIM, DIM);
    cvtT(W1   + (size_t)l * DIM * HIDD, w1_bf, DIM, HIDD);
    cvtT(W2   + (size_t)l * HIDD * DIM, w2_bf, HIDD, DIM);

    // xm = modulate(ln(h), sh_a, sc_a)
    ln_mod_kernel<<<MM, blk, 0, stream>>>(h, mods, 0 * DIM, 1 * DIM, 6 * DIM, xm);
    // qkv = xm @ Wqkv + bqkv
    gemm_bf16_kernel<<<dim3(QKVS / 64, MM / 64), blk, 0, stream>>>(
        xm, wq_bf, bqkv + (size_t)l * QKVS, qkvb, nullptr, nullptr,
        DIM, QKVS, 0, EPI_STORE);
    // attention
    attention_kernel<<<dim3(BB * NH, SEQ / 128), blk, 0, stream>>>(qkvb, mask, attnb);
    // h += g_a * (o @ Wo + bo)
    gemm_bf16_kernel<<<dim3(DIM / 64, MM / 64), blk, 0, stream>>>(
        attnb, wo_bf, bo + (size_t)l * DIM, nullptr, h, mods + 2 * DIM,
        DIM, DIM, 6 * DIM, EPI_GATE);
    // y = modulate(ln(h), sh_m, sc_m)
    ln_mod_kernel<<<MM, blk, 0, stream>>>(h, mods, 3 * DIM, 4 * DIM, 6 * DIM, xm);
    // hid = gelu(y @ W1 + b1)
    gemm_bf16_kernel<<<dim3(HIDD / 64, MM / 64), blk, 0, stream>>>(
        xm, w1_bf, b1 + (size_t)l * HIDD, hidb, nullptr, nullptr,
        DIM, HIDD, 0, EPI_GELU);
    // h += g_m * (hid @ W2 + b2)
    gemm_bf16_kernel<<<dim3(DIM / 64, MM / 64), blk, 0, stream>>>(
        hidb, w2_bf, b2 + (size_t)l * DIM, nullptr, h, mods + 5 * DIM,
        HIDD, DIM, 6 * DIM, EPI_GATE);
  }

  ada_kernel<<<dim3((2 * DIM) / 256, BB), blk, 0, stream>>>(
      csilu, Wada_f, bada_f, modsf, 2 * DIM);
  final_kernel<<<MM, blk, 0, stream>>>(h, modsf, W_out, out);
}
